// LATTE_80917183857182
// MI455X (gfx1250) — matrix operand order repaired and checked by reading, not compiled, not run on hardware
//
#include <hip/hip_runtime.h>
#include <hip/hip_bf16.h>

// ---------------------------------------------------------------------------
// CDNA5 (gfx1250) implementation of a 2-node-type graph-attention layer.
// Dense projections: register-blocked bf16 WMMA GEMM (v_wmma_f32_16x16x32_bf16,
// fp32 accumulate), one wave per 16-row stripe covering the full N dimension
// (8 resident accumulators) so A is loaded once per K-step and 8 WMMAs issue
// back-to-back. Edge softmax/aggregation is gather/atomic bound (wave32).
// ---------------------------------------------------------------------------

typedef __attribute__((ext_vector_type(16))) __bf16 v16bf;
typedef __attribute__((ext_vector_type(8)))  float  v8f;

#define D_EMB 128
#define C_HEAD 32
#define F_IN 256

// ---------------- utility ----------------

__global__ void fill_f32(float* __restrict__ p, float v, int n) {
    int i = blockIdx.x * blockDim.x + threadIdx.x;
    if (i < n) p[i] = v;
}

__device__ __forceinline__ float wave_reduce_sum(float v) {
#pragma unroll
    for (int off = 16; off > 0; off >>= 1) v += __shfl_down(v, off, 32);
    return __shfl(v, 0, 32);
}

// float atomic max via monotone int mapping (dst initialized to -inf)
__device__ __forceinline__ void atomicMaxFloat(float* addr, float val) {
    if (val >= 0.0f)
        atomicMax((int*)addr, __float_as_int(val));
    else
        atomicMin((unsigned int*)addr, __float_as_uint(val));
}

// ---------------- WMMA GEMM:  Y[M,N] = X[M,K] @ W[N,K]^T + b[N] ----------------
// One wave32 per 16-row M stripe; NTN 16-col tiles held in registers
// (NTN v8f accumulators). A (16x32 bf16) loaded once per K-step, then NTN
// back-to-back v_wmma_f32_16x16x32_bf16. Requires M%16==0, N==NTN*16, K%32==0.
template <int NTN>
__global__ void gemm_bf16_wmma(const float* __restrict__ X,
                               const float* __restrict__ W,
                               const float* __restrict__ bias,
                               float* __restrict__ Y,
                               int M, int K) {
    constexpr int N = NTN * 16;
    int gid  = blockIdx.x * blockDim.x + threadIdx.x;
    int wave = gid >> 5;
    int lane = gid & 31;
    if (wave >= (M >> 4)) return;

    int h  = lane >> 4;   // half-wave (0/1)
    int mr = lane & 15;   // A row within stripe / B column within tile

    v8f acc[NTN];
#pragma unroll
    for (int t = 0; t < NTN; ++t) acc[t] = (v8f){};

    const float* xrow = X + (size_t)(wave * 16 + mr) * K;

    for (int k0 = 0; k0 < K; k0 += 32) {
        // A: 16x32 bf16. elements e -> K = 16*h + e
        v16bf a;
        const float4* xa0 = (const float4*)(xrow + k0 + h * 16);
        const float4* xa1 = (const float4*)(xrow + k0 + h * 16 + 8);
        float4 x0 = xa0[0], x1 = xa0[1], x2 = xa1[0], x3 = xa1[1];
        a[0]  = (__bf16)x0.x; a[1]  = (__bf16)x0.y; a[2]  = (__bf16)x0.z; a[3]  = (__bf16)x0.w;
        a[4]  = (__bf16)x1.x; a[5]  = (__bf16)x1.y; a[6]  = (__bf16)x1.z; a[7]  = (__bf16)x1.w;
        a[8]  = (__bf16)x2.x; a[9]  = (__bf16)x2.y; a[10] = (__bf16)x2.z; a[11] = (__bf16)x2.w;
        a[12] = (__bf16)x3.x; a[13] = (__bf16)x3.y; a[14] = (__bf16)x3.z; a[15] = (__bf16)x3.w;

#pragma unroll
        for (int t = 0; t < NTN; ++t) {
            // B: 32x16 bf16, B[k][n] = W[n][k]; lane holds col n = t*16+mr,
            // elements e -> K = 16*h + e : 16 contiguous floats of W's row.
            v16bf b;
            const float4* wb = (const float4*)(W + (size_t)(t * 16 + mr) * K + k0 + h * 16);
            float4 w0 = wb[0], w1 = wb[1], w2 = wb[2], w3 = wb[3];
            b[0]  = (__bf16)w0.x; b[1]  = (__bf16)w0.y; b[2]  = (__bf16)w0.z; b[3]  = (__bf16)w0.w;
            b[4]  = (__bf16)w1.x; b[5]  = (__bf16)w1.y; b[6]  = (__bf16)w1.z; b[7]  = (__bf16)w1.w;
            b[8]  = (__bf16)w2.x; b[9]  = (__bf16)w2.y; b[10] = (__bf16)w2.z; b[11] = (__bf16)w2.w;
            b[12] = (__bf16)w3.x; b[13] = (__bf16)w3.y; b[14] = (__bf16)w3.z; b[15] = (__bf16)w3.w;

            acc[t] = __builtin_amdgcn_wmma_f32_16x16x32_bf16(
                /*neg_a=*/false, a, /*neg_b=*/false, b,
                /*c_mod=*/(short)0, acc[t], /*reuse_a=*/false, /*reuse_b=*/false);
        }
    }

    // D layout: element r -> row wave*16 + r + 8*h, col t*16 + mr
#pragma unroll
    for (int t = 0; t < NTN; ++t) {
        int col = t * 16 + mr;
        float bv = bias[col];
#pragma unroll
        for (int r = 0; r < 8; ++r) {
            int row = wave * 16 + r + 8 * h;
            Y[(size_t)row * N + col] = acc[t][r] + bv;
        }
    }
}

// ---------------- edge phase ----------------

// one wave per edge: e = (tanh([a_i, a_j]) . qw) * sharp ; atomic segment max
__global__ void edge_logit_max(const float* __restrict__ al,
                               const float* __restrict__ ar,
                               const int* __restrict__ eidx,   // [2,E]
                               const float* __restrict__ qw,   // [64]
                               const float* __restrict__ sharp,
                               float* __restrict__ elog,       // [E]
                               float* __restrict__ segmax,     // [n_head], init -inf
                               int E) {
    int gid  = blockIdx.x * blockDim.x + threadIdx.x;
    int wave = gid >> 5;
    int lane = gid & 31;
    if (wave >= E) return;
    int src = eidx[wave];       // head (aggregate-at)
    int dst = eidx[E + wave];   // tail (message-from)
    float v = tanhf(al[(size_t)src * C_HEAD + lane]) * qw[lane]
            + tanhf(ar[(size_t)dst * C_HEAD + lane]) * qw[C_HEAD + lane];
    v = wave_reduce_sum(v);
    if (lane == 0) {
        float e = v * sharp[0];
        elog[wave] = e;
        atomicMaxFloat(segmax + src, e);
    }
}

// one thread per edge: ex = exp(e - m[src]); denom[src] += ex
__global__ void edge_expsum(const int* __restrict__ eidx,
                            const float* __restrict__ segmax,
                            float* __restrict__ elog,
                            float* __restrict__ denom,  // init 0
                            int E) {
    int e = blockIdx.x * blockDim.x + threadIdx.x;
    if (e >= E) return;
    int src = eidx[e];
    float ex = expf(elog[e] - segmax[src]);
    elog[e] = ex;
    atomicAdd(denom + src, ex);
}

// one wave per edge: agg[src] += alpha * r_tail[dst]   (128 channels, 4/lane)
__global__ void edge_scatter(const int* __restrict__ eidx,
                             const float* __restrict__ r_tail,
                             const float* __restrict__ elog,
                             const float* __restrict__ denom,
                             float* __restrict__ agg,   // init 0
                             int E) {
    int gid  = blockIdx.x * blockDim.x + threadIdx.x;
    int wave = gid >> 5;
    int lane = gid & 31;
    if (wave >= E) return;
    int src = eidx[wave];
    int dst = eidx[E + wave];
    float alpha = elog[wave] / (denom[src] + 1e-16f);
    const float* rrow = r_tail + (size_t)dst * D_EMB;
    float* arow = agg + (size_t)src * D_EMB;
#pragma unroll
    for (int i = 0; i < 4; ++i) {
        int d = lane + 32 * i;
        atomicAdd(arow + d, rrow[d] * alpha);
    }
}

// ---------------- relation combine ----------------
// rows = [aggA, (aggB), l_self]; beta = softmax(rows . convW + b); out=relu(sum beta*rows)
// One wave per node. Safe when out aliases aggA (rows read to regs first).
__global__ void combine_kernel(const float* __restrict__ aggA,
                               const float* __restrict__ aggB,
                               const float* __restrict__ lself,
                               const float* __restrict__ convW,  // [128]
                               const float* __restrict__ convb,  // [1]
                               float* __restrict__ out,
                               int N, int hasB) {
    int gid  = blockIdx.x * blockDim.x + threadIdx.x;
    int wave = gid >> 5;
    int lane = gid & 31;
    if (wave >= N) return;
    int R = hasB ? 3 : 2;

    const float* ps[3];
    ps[0] = aggA;
    ps[1] = hasB ? aggB : lself;
    ps[2] = lself;

    float wv[4];
#pragma unroll
    for (int i = 0; i < 4; ++i) wv[i] = convW[lane + 32 * i];

    float rows[3][4];
    float s[3];
    for (int r = 0; r < R; ++r) {
        const float* p = ps[r] + (size_t)wave * D_EMB;
        float dot = 0.0f;
#pragma unroll
        for (int i = 0; i < 4; ++i) {
            float x = p[lane + 32 * i];
            rows[r][i] = x;
            dot += x * wv[i];
        }
        s[r] = wave_reduce_sum(dot) + convb[0];
    }
    float mx = s[0];
    for (int r = 1; r < R; ++r) mx = fmaxf(mx, s[r]);
    float b[3]; float bsum = 0.0f;
    for (int r = 0; r < R; ++r) { b[r] = expf(s[r] - mx); bsum += b[r]; }
#pragma unroll
    for (int i = 0; i < 4; ++i) {
        float acc = 0.0f;
        for (int r = 0; r < R; ++r) acc += rows[r][i] * (b[r] / bsum);
        out[(size_t)wave * D_EMB + lane + 32 * i] = fmaxf(acc, 0.0f);
    }
}

// ---------------- host orchestration ----------------

static inline int cdiv(long long a, long long b) { return (int)((a + b - 1) / b); }

extern "C" void kernel_launch(void* const* d_in, const int* in_sizes, int n_in,
                              void* d_out, int out_size, void* d_ws, size_t ws_size,
                              hipStream_t stream) {
    const float* x_gene   = (const float*)d_in[0];
    const float* x_prot   = (const float*)d_in[1];
    const float* Wl_gene  = (const float*)d_in[2];
    const float* bl_gene  = (const float*)d_in[3];
    const float* Wr_gene  = (const float*)d_in[4];
    const float* br_gene  = (const float*)d_in[5];
    const float* Wl_prot  = (const float*)d_in[6];
    const float* bl_prot  = (const float*)d_in[7];
    const float* Wr_prot  = (const float*)d_in[8];
    const float* br_prot  = (const float*)d_in[9];
    const float* alW      = (const float*)d_in[10];  // [3,32,128]
    const float* alb      = (const float*)d_in[11];  // [3,32]
    const float* arW      = (const float*)d_in[12];
    const float* arb      = (const float*)d_in[13];
    const float* qw       = (const float*)d_in[14];  // [3,64]
    const float* sharp    = (const float*)d_in[15];  // [3]
    const float* convgW   = (const float*)d_in[16];
    const float* convgb   = (const float*)d_in[17];
    const float* convpW   = (const float*)d_in[18];
    const float* convpb   = (const float*)d_in[19];
    const int*   edge_gg  = (const int*)d_in[20];
    const int*   edge_gp  = (const int*)d_in[21];
    const int*   edge_pp  = (const int*)d_in[22];

    const int NG = in_sizes[0] / F_IN;     // 50000
    const int NP = in_sizes[1] / F_IN;     // 50000
    const int E  = in_sizes[20] / 2;       // 500000

    // ---- workspace carve (floats) ----
    float* w = (float*)d_ws;
    float* l_gene = w;  w += (size_t)NG * D_EMB;
    float* r_gene = w;  w += (size_t)NG * D_EMB;
    float* l_prot = w;  w += (size_t)NP * D_EMB;
    float* r_prot = w;  w += (size_t)NP * D_EMB;
    float* al_g0  = w;  w += (size_t)NG * C_HEAD;
    float* ar_g0  = w;  w += (size_t)NG * C_HEAD;
    float* al_g1  = w;  w += (size_t)NG * C_HEAD;
    float* ar_p1  = w;  w += (size_t)NP * C_HEAD;
    float* al_p2  = w;  w += (size_t)NP * C_HEAD;
    float* ar_p2  = w;  w += (size_t)NP * C_HEAD;
    float* agg_gp = w;  w += (size_t)NG * D_EMB;
    float* elog   = w;  w += (size_t)E;
    float* segmax = w;  w += (size_t)(NG > NP ? NG : NP);
    float* denom  = w;  w += (size_t)(NG > NP ? NG : NP);

    float* out_gene = (float*)d_out;
    float* out_prot = out_gene + (size_t)NG * D_EMB;
    float* agg_gg = out_gene;   // aliased: combine reads rows before writing
    float* agg_pp = out_prot;

    // ---- 1) node projections (WMMA GEMMs), K=256, N=128 (8 N-tiles/wave) ----
    {
        int blocks = cdiv((long long)(NG / 16) * 32, 128);
        gemm_bf16_wmma<8><<<blocks, 128, 0, stream>>>(x_gene, Wl_gene, bl_gene, l_gene, NG, F_IN);
        gemm_bf16_wmma<8><<<blocks, 128, 0, stream>>>(x_gene, Wr_gene, br_gene, r_gene, NG, F_IN);
        gemm_bf16_wmma<8><<<blocks, 128, 0, stream>>>(x_prot, Wl_prot, bl_prot, l_prot, NP, F_IN);
        gemm_bf16_wmma<8><<<blocks, 128, 0, stream>>>(x_prot, Wr_prot, br_prot, r_prot, NP, F_IN);
    }
    // ---- 2) attention projections (WMMA), K=128, N=32 (2 N-tiles/wave) ----
    {
        int blocks = cdiv((long long)(NG / 16) * 32, 128);
        const int WS = C_HEAD * D_EMB;  // 4096 per metapath
        gemm_bf16_wmma<2><<<blocks, 128, 0, stream>>>(l_gene, alW + 0 * WS, alb + 0,  al_g0, NG, D_EMB);
        gemm_bf16_wmma<2><<<blocks, 128, 0, stream>>>(r_gene, arW + 0 * WS, arb + 0,  ar_g0, NG, D_EMB);
        gemm_bf16_wmma<2><<<blocks, 128, 0, stream>>>(l_gene, alW + 1 * WS, alb + 32, al_g1, NG, D_EMB);
        gemm_bf16_wmma<2><<<blocks, 128, 0, stream>>>(r_prot, arW + 1 * WS, arb + 32, ar_p1, NP, D_EMB);
        gemm_bf16_wmma<2><<<blocks, 128, 0, stream>>>(l_prot, alW + 2 * WS, alb + 64, al_p2, NP, D_EMB);
        gemm_bf16_wmma<2><<<blocks, 128, 0, stream>>>(r_prot, arW + 2 * WS, arb + 64, ar_p2, NP, D_EMB);
    }

    // ---- 3) zero aggregation targets ----
    fill_f32<<<cdiv((long long)NG * D_EMB, 256), 256, 0, stream>>>(agg_gg, 0.0f, NG * D_EMB);
    fill_f32<<<cdiv((long long)NG * D_EMB, 256), 256, 0, stream>>>(agg_gp, 0.0f, NG * D_EMB);
    fill_f32<<<cdiv((long long)NP * D_EMB, 256), 256, 0, stream>>>(agg_pp, 0.0f, NP * D_EMB);

    // ---- 4) metapath aggregation (sequential; segmax/denom/elog reused) ----
    struct MP { const float* al; const float* ar; const float* rt;
                const int* e; const float* q; const float* sh;
                float* agg; int nh; };
    const MP mps[3] = {
        { al_g0, ar_g0, r_gene, edge_gg, qw + 0,   sharp + 0, agg_gg, NG },
        { al_g1, ar_p1, r_prot, edge_gp, qw + 64,  sharp + 1, agg_gp, NG },
        { al_p2, ar_p2, r_prot, edge_pp, qw + 128, sharp + 2, agg_pp, NP },
    };
    const int ewave_blocks = cdiv((long long)E * 32, 256);
    const int e_blocks     = cdiv((long long)E, 256);
    for (int m = 0; m < 3; ++m) {
        const MP& p = mps[m];
        fill_f32<<<cdiv(p.nh, 256), 256, 0, stream>>>(segmax, -INFINITY, p.nh);
        fill_f32<<<cdiv(p.nh, 256), 256, 0, stream>>>(denom, 0.0f, p.nh);
        edge_logit_max<<<ewave_blocks, 256, 0, stream>>>(p.al, p.ar, p.e, p.q, p.sh, elog, segmax, E);
        edge_expsum<<<e_blocks, 256, 0, stream>>>(p.e, segmax, elog, denom, E);
        edge_scatter<<<ewave_blocks, 256, 0, stream>>>(p.e, p.rt, elog, denom, p.agg, E);
    }

    // ---- 5) relation combine ----
    combine_kernel<<<cdiv((long long)NG * 32, 256), 256, 0, stream>>>(
        agg_gg, agg_gp, l_gene, convgW, convgb, out_gene, NG, 1);
    combine_kernel<<<cdiv((long long)NP * 32, 256), 256, 0, stream>>>(
        agg_pp, nullptr, l_prot, convpW, convpb, out_prot, NP, 0);
}
